// InputBlock_76785425318091
// MI455X (gfx1250) — compile-verified, matrix operand order checked by reading, not run on hardware
//
#include <hip/hip_runtime.h>
#include <hip/hip_bf16.h>
#include <math.h>

typedef __attribute__((ext_vector_type(16))) __bf16 v16bf;
typedef __attribute__((ext_vector_type(8)))  float  v8f;

#define NNODES 10000
#define KNEI   32
#define FDIM   128
#define HDIM   256
#define NROWS  (NNODES * KNEI)   // 320000
#define EPSV   1e-5f

// ---- workspace layout (bytes): packed bf16 weights only ----
#define WS_WLIN 0                               // bf16 [4*16*512]  = 64 KB
#define WS_WFF1 (65536)                         // bf16 [8*16*512]  = 128 KB
#define WS_WFF2 (65536 + 131072)                // bf16 [8*16*512]  = 128 KB

// K index inside a 32-wide K-tile for element e of lane l (16-bit WMMA A/B layout):
// lanes 0-15: e0..7 -> K 0..7,  e8..15 -> K 16..23
// lanes16-31: e0..7 -> K 8..15, e8..15 -> K 24..31
__device__ __host__ __forceinline__ int kmap(int e, int lane) {
    return e + ((e >= 8) ? 8 : 0) + ((lane >= 16) ? 8 : 0);
}

__device__ __forceinline__ float gelu_fast(float x) {
    const float c0 = 0.7978845608028654f;
    const float c1 = 0.044715f;
    float u = c0 * (x + c1 * x * x * x);
    // 0.5*x*(1+tanh(u)) == x * sigmoid(2u)
    float e = __expf(-2.0f * u);                  // v_exp_f32
    return x * __builtin_amdgcn_rcpf(1.0f + e);   // v_rcp_f32
}

__device__ __forceinline__ v16bf make_af(float4 a0, float4 a1, float4 a2, float4 a3) {
    v16bf v;
    v[0]  = (__bf16)a0.x; v[1]  = (__bf16)a0.y; v[2]  = (__bf16)a0.z; v[3]  = (__bf16)a0.w;
    v[4]  = (__bf16)a1.x; v[5]  = (__bf16)a1.y; v[6]  = (__bf16)a1.z; v[7]  = (__bf16)a1.w;
    v[8]  = (__bf16)a2.x; v[9]  = (__bf16)a2.y; v[10] = (__bf16)a2.z; v[11] = (__bf16)a2.w;
    v[12] = (__bf16)a3.x; v[13] = (__bf16)a3.y; v[14] = (__bf16)a3.z; v[15] = (__bf16)a3.w;
    return v;
}

// ---- kernel 1: pack f32 weights [Kdim x 256] into bf16 WMMA B-fragment order ----
// frag layout: [kTile][nTile][lane 0..31][e 0..15], element = W[kTile*32 + kmap(e,l), nTile*16 + (l&15)]
__global__ void pack_w(const float* __restrict__ W, __bf16* __restrict__ out, int Kdim) {
    int i = blockIdx.x * blockDim.x + threadIdx.x;
    int total = Kdim * HDIM;
    if (i >= total) return;
    int frag   = i >> 9;      // /512 elements per fragment
    int within = i & 511;
    int lane   = within >> 4;
    int e      = within & 15;
    int nt = frag & 15;
    int kt = frag >> 4;
    int k = kt * 32 + kmap(e, lane);
    int n = nt * 16 + (lane & 15);
    out[i] = (__bf16)W[(size_t)k * HDIM + n];
}

// ---- kernel 2: fused edge FFN + edge LN + masked node aggregation + node LN ----
// Block = 128 threads (4 waves), 64 rows = 2 nodes. Wave w owns rows [w*16, w*16+16).
__global__ __launch_bounds__(128)
void fused_block(const float* __restrict__ edge_features,
                 const unsigned char* __restrict__ mask,
                 const __bf16* __restrict__ wlin,
                 const __bf16* __restrict__ wff1,
                 const __bf16* __restrict__ wff2,
                 const float* __restrict__ b_lin, const float* __restrict__ g_pre,
                 const float* __restrict__ b_pre, const float* __restrict__ b_ff1,
                 const float* __restrict__ b_ff2,
                 const float* __restrict__ g_node, const float* __restrict__ b_node,
                 const float* __restrict__ g_edge, const float* __restrict__ b_edge,
                 float* __restrict__ node_out,        // [NNODES][HDIM]
                 float* __restrict__ edge_out)        // [NROWS][HDIM] (normalized)
{
    __shared__ float  H1[4][16][HDIM];    // edge_hidden / edge_output per wave
    __shared__ float  H2[4][16][HDIM];    // normalized / gelu activations per wave
    __shared__ float4 part4[4][64];       // per-wave masked partial sums (256 ch as 64 quads)
    __shared__ float  rowMean[4][16];     // edge_output row stats
    __shared__ float  rowRstd[4][16];
    __shared__ float  red[128];           // node LN reduction scratch

    const int lane = threadIdx.x & 31;
    const int w    = threadIdx.x >> 5;
    const int hi   = lane >> 4;        // 0 or 1 (lane half)
    const int lm   = lane & 15;
    const int rowBase = blockIdx.x * 64 + w * 16;

    v8f acc[16];
    const v8f vzero = {0.f, 0.f, 0.f, 0.f, 0.f, 0.f, 0.f, 0.f};

    // ================= GEMM1: edge_hidden = A[16x128] * W_lin[128x256] =================
#pragma unroll
    for (int nt = 0; nt < 16; ++nt) acc[nt] = vzero;
    {
        const float* rowPtr = edge_features + (size_t)(rowBase + lm) * FDIM;
        for (int kt = 0; kt < 4; ++kt) {
            int ofs = kt * 32 + hi * 8;
            float4 a0 = *(const float4*)(rowPtr + ofs);
            float4 a1 = *(const float4*)(rowPtr + ofs + 4);
            float4 a2 = *(const float4*)(rowPtr + ofs + 16);
            float4 a3 = *(const float4*)(rowPtr + ofs + 20);
            v16bf af = make_af(a0, a1, a2, a3);
#pragma unroll
            for (int nt = 0; nt < 16; ++nt) {
                v16bf bf = *(const v16bf*)(wlin + ((size_t)(kt * 16 + nt) * 512 + lane * 16));
                acc[nt] = __builtin_amdgcn_wmma_f32_16x16x32_bf16(
                    false, af, false, bf, (short)0, acc[nt], false, false);
            }
        }
    }

    // bias + stash edge_hidden into H1
#pragma unroll
    for (int nt = 0; nt < 16; ++nt) {
        float bl = b_lin[nt * 16 + lm];
#pragma unroll
        for (int r = 0; r < 8; ++r) {
            H1[w][r + hi * 8][nt * 16 + lm] = acc[nt][r] + bl;
        }
    }
    __syncthreads();

    // ================= pre-norm layernorm: H2 = LN(H1)*g_pre + b_pre (float4 sweeps) ==========
    {
        int row = lane >> 1, half = lane & 1;
        const float4* src = (const float4*)&H1[w][row][half * 128];   // 32 quads
        float s = 0.f;
#pragma unroll 8
        for (int i = 0; i < 32; ++i) { float4 v = src[i]; s += v.x + v.y + v.z + v.w; }
        s += __shfl_xor(s, 1, 32);
        float mean = s * (1.f / 256.f);
        float vv = 0.f;
#pragma unroll 8
        for (int i = 0; i < 32; ++i) {
            float4 v = src[i];
            float dx = v.x - mean, dy = v.y - mean, dz = v.z - mean, dw = v.w - mean;
            vv += dx * dx + dy * dy + dz * dz + dw * dw;
        }
        vv += __shfl_xor(vv, 1, 32);
        float rstd = rsqrtf(vv * (1.f / 256.f) + EPSV);
        float4* dst = (float4*)&H2[w][row][half * 128];
#pragma unroll 8
        for (int i = 0; i < 32; ++i) {
            int c = half * 128 + i * 4;
            float4 v = src[i];
            float4 g = *(const float4*)(g_pre + c);
            float4 b = *(const float4*)(b_pre + c);
            float4 o;
            o.x = (v.x - mean) * rstd * g.x + b.x;
            o.y = (v.y - mean) * rstd * g.y + b.y;
            o.z = (v.z - mean) * rstd * g.z + b.z;
            o.w = (v.w - mean) * rstd * g.w + b.w;
            dst[i] = o;
        }
    }
    __syncthreads();

    // ================= GEMM2: h1 = H2[16x256] * W_ff1[256x256] =================
#pragma unroll
    for (int nt = 0; nt < 16; ++nt) acc[nt] = vzero;
    for (int kt = 0; kt < 8; ++kt) {
        const float* arow = &H2[w][lm][kt * 32 + hi * 8];
        float4 a0 = *(const float4*)(arow);
        float4 a1 = *(const float4*)(arow + 4);
        float4 a2 = *(const float4*)(arow + 16);
        float4 a3 = *(const float4*)(arow + 20);
        v16bf af = make_af(a0, a1, a2, a3);
#pragma unroll
        for (int nt = 0; nt < 16; ++nt) {
            v16bf bf = *(const v16bf*)(wff1 + ((size_t)(kt * 16 + nt) * 512 + lane * 16));
            acc[nt] = __builtin_amdgcn_wmma_f32_16x16x32_bf16(
                false, af, false, bf, (short)0, acc[nt], false, false);
        }
    }
    __syncthreads();

    // bias + gelu -> overwrite H2
#pragma unroll
    for (int nt = 0; nt < 16; ++nt) {
        float b1 = b_ff1[nt * 16 + lm];
#pragma unroll
        for (int r = 0; r < 8; ++r) {
            H2[w][r + hi * 8][nt * 16 + lm] = gelu_fast(acc[nt][r] + b1);
        }
    }
    __syncthreads();

    // ================= GEMM3: h2 = H2[16x256] * W_ff2[256x256] =================
#pragma unroll
    for (int nt = 0; nt < 16; ++nt) acc[nt] = vzero;
    for (int kt = 0; kt < 8; ++kt) {
        const float* arow = &H2[w][lm][kt * 32 + hi * 8];
        float4 a0 = *(const float4*)(arow);
        float4 a1 = *(const float4*)(arow + 4);
        float4 a2 = *(const float4*)(arow + 16);
        float4 a3 = *(const float4*)(arow + 20);
        v16bf af = make_af(a0, a1, a2, a3);
#pragma unroll
        for (int nt = 0; nt < 16; ++nt) {
            v16bf bf = *(const v16bf*)(wff2 + ((size_t)(kt * 16 + nt) * 512 + lane * 16));
            acc[nt] = __builtin_amdgcn_wmma_f32_16x16x32_bf16(
                false, af, false, bf, (short)0, acc[nt], false, false);
        }
    }
    __syncthreads();

    // bias + residual -> edge_output into H1
#pragma unroll
    for (int nt = 0; nt < 16; ++nt) {
        float b2 = b_ff2[nt * 16 + lm];
#pragma unroll
        for (int r = 0; r < 8; ++r) {
            int m = r + hi * 8, c = nt * 16 + lm;
            H1[w][m][c] = acc[nt][r] + b2 + H1[w][m][c];
        }
    }
    __syncthreads();

    // ---- edge_output row stats (for edge layernorm) ----
    {
        int row = lane >> 1, half = lane & 1;
        const float4* src = (const float4*)&H1[w][row][half * 128];
        float s = 0.f;
#pragma unroll 8
        for (int i = 0; i < 32; ++i) { float4 v = src[i]; s += v.x + v.y + v.z + v.w; }
        s += __shfl_xor(s, 1, 32);
        float mean = s * (1.f / 256.f);
        float vv = 0.f;
#pragma unroll 8
        for (int i = 0; i < 32; ++i) {
            float4 v = src[i];
            float dx = v.x - mean, dy = v.y - mean, dz = v.z - mean, dw = v.w - mean;
            vv += dx * dx + dy * dy + dz * dz + dw * dw;
        }
        vv += __shfl_xor(vv, 1, 32);
        float rstd = rsqrtf(vv * (1.f / 256.f) + EPSV);
        if (half == 0) {
            rowMean[w][row] = mean;
            rowRstd[w][row] = rstd;
        }
    }

    // ---- per-wave masked partial sums over this wave's 16 rows ----
    {
        float mkf[16];
#pragma unroll
        for (int m = 0; m < 16; ++m) mkf[m] = (float)mask[rowBase + m];
        for (int q = lane; q < 64; q += 32) {
            float4 s = {0.f, 0.f, 0.f, 0.f};
#pragma unroll
            for (int m = 0; m < 16; ++m) {
                float4 v = *(const float4*)(&H1[w][m][q * 4]);
                s.x += v.x * mkf[m]; s.y += v.y * mkf[m];
                s.z += v.z * mkf[m]; s.w += v.w * mkf[m];
            }
            part4[w][q] = s;
        }
    }
    __syncthreads();

    // ---- node aggregation + node layernorm (2 nodes per block) ----
    {
        int t = threadIdx.x;           // 0..127
        int j = t >> 6;                // node within block (0/1): waves 2j, 2j+1
        int q = t & 63;                // channel quad
        float4 pa = part4[2 * j][q];
        float4 pb = part4[2 * j + 1][q];
        float4 s;
        s.x = pa.x + pb.x; s.y = pa.y + pb.y; s.z = pa.z + pb.z; s.w = pa.w + pb.w;

        red[t] = s.x + s.y + s.z + s.w;
        __syncthreads();
        for (int st = 32; st > 0; st >>= 1) { if (q < st) red[t] += red[t + st]; __syncthreads(); }
        float mean = red[j * 64] * (1.f / 256.f);
        __syncthreads();
        float dx = s.x - mean, dy = s.y - mean, dz = s.z - mean, dw = s.w - mean;
        red[t] = dx * dx + dy * dy + dz * dz + dw * dw;
        __syncthreads();
        for (int st = 32; st > 0; st >>= 1) { if (q < st) red[t] += red[t + st]; __syncthreads(); }
        float rstd = rsqrtf(red[j * 64] * (1.f / 256.f) + EPSV);

        int c = q * 4;
        float4 g = *(const float4*)(g_node + c);
        float4 b = *(const float4*)(b_node + c);
        float4 o;
        o.x = dx * rstd * g.x + b.x;
        o.y = dy * rstd * g.y + b.y;
        o.z = dz * rstd * g.z + b.z;
        o.w = dw * rstd * g.w + b.w;
        *(float4*)(node_out + (size_t)(blockIdx.x * 2 + j) * HDIM + c) = o;
    }

    // ---- normalized edge output write (coalesced, LN applied in flight) ----
    for (int idx = lane; idx < 16 * 64; idx += 32) {
        int rr = idx >> 6;
        int c4 = (idx & 63) * 4;
        float4 v  = *(const float4*)(&H1[w][rr][c4]);
        float mean = rowMean[w][rr];
        float rstd = rowRstd[w][rr];
        float4 g = *(const float4*)(g_edge + c4);
        float4 b = *(const float4*)(b_edge + c4);
        float4 o;
        o.x = (v.x - mean) * rstd * g.x + b.x;
        o.y = (v.y - mean) * rstd * g.y + b.y;
        o.z = (v.z - mean) * rstd * g.z + b.z;
        o.w = (v.w - mean) * rstd * g.w + b.w;
        *(float4*)(edge_out + (size_t)(rowBase + rr) * HDIM + c4) = o;
    }
}

extern "C" void kernel_launch(void* const* d_in, const int* in_sizes, int n_in,
                              void* d_out, int out_size, void* d_ws, size_t ws_size,
                              hipStream_t stream)
{
    (void)in_sizes; (void)n_in; (void)out_size; (void)ws_size;

    const float*         edge_features = (const float*)d_in[0];
    const unsigned char* mask          = (const unsigned char*)d_in[1];
    const float* W_lin  = (const float*)d_in[2];
    const float* b_lin  = (const float*)d_in[3];
    const float* g_pre  = (const float*)d_in[4];
    const float* b_pre  = (const float*)d_in[5];
    const float* W_ff1  = (const float*)d_in[6];
    const float* b_ff1  = (const float*)d_in[7];
    const float* W_ff2  = (const float*)d_in[8];
    const float* b_ff2  = (const float*)d_in[9];
    const float* g_node = (const float*)d_in[10];
    const float* b_node = (const float*)d_in[11];
    const float* g_edge = (const float*)d_in[12];
    const float* b_edge = (const float*)d_in[13];

    char* ws = (char*)d_ws;
    __bf16* wlin = (__bf16*)(ws + WS_WLIN);
    __bf16* wff1 = (__bf16*)(ws + WS_WFF1);
    __bf16* wff2 = (__bf16*)(ws + WS_WFF2);

    float* node_out = (float*)d_out;                              // [NNODES][HDIM]
    float* edge_out = (float*)d_out + (size_t)NNODES * HDIM;      // [NROWS][HDIM]

    pack_w<<<(FDIM * HDIM + 255) / 256, 256, 0, stream>>>(W_lin, wlin, FDIM);
    pack_w<<<(HDIM * HDIM + 255) / 256, 256, 0, stream>>>(W_ff1, wff1, HDIM);
    pack_w<<<(HDIM * HDIM + 255) / 256, 256, 0, stream>>>(W_ff2, wff2, HDIM);

    fused_block<<<NROWS / 64, 128, 0, stream>>>(edge_features, mask,
                                                wlin, wff1, wff2,
                                                b_lin, g_pre, b_pre, b_ff1, b_ff2,
                                                g_node, b_node, g_edge, b_edge,
                                                node_out, edge_out);
}
